// Decoder_42099269435401
// MI455X (gfx1250) — compile-verified, matrix operand order checked by reading, not run on hardware
//
#include <hip/hip_runtime.h>
#include <math.h>

// ---------------------------------------------------------------------------
// Decoder (B=2, S=SE=2048, D=1024, H=16, dh=64, L=4) for gfx1250 (CDNA5).
// bf16 data path end-to-end: weights pre-transposed to bf16 [N][K], all
// activations shadowed in bf16 by producer epilogues. GEMM hot loop is pure
// async global->LDS b128 copies (ASYNCcnt, double buffered) + wmma_bf16.
// Attention is fused flash-style with per-head pre-laid-out bf16 q/k/v.
// Workspace: ~152 MB.
// ---------------------------------------------------------------------------

#define D_MODEL 1024
#define NHEAD   16
#define DH      64
#define SEQ     2048
#define SENC    2048
#define BATCH   2
#define NLAYER  4
#define NTOK    (BATCH * SEQ)          // 4096 rows

typedef unsigned short ushort_t;
typedef __attribute__((ext_vector_type(16))) __bf16 v16bf;
typedef __attribute__((ext_vector_type(8)))  float  v8f;

union FragAB { uint4 q[2]; v16bf v; };

static __device__ __forceinline__ unsigned short f2bf(float f) {
  unsigned int u = __float_as_uint(f);
  u += 0x7fffu + ((u >> 16) & 1u);     // round-to-nearest-even
  return (unsigned short)(u >> 16);
}

// Async 16-byte global -> LDS copy (per-lane addresses, tracked by ASYNCcnt).
static __device__ __forceinline__ void async_cp16(unsigned lds_byte_addr,
                                                  const void* gptr) {
  asm volatile("global_load_async_to_lds_b128 %0, %1, off"
               :: "v"(lds_byte_addr),
                  "v"((unsigned long long)(uintptr_t)gptr)
               : "memory");
}
static __device__ __forceinline__ void wait_async0() {
  asm volatile("s_wait_asynccnt 0x0" ::: "memory");
}

// Epilogue modes
#define MODE_F32         0   // Cf[m][n] = acc + bias               (fp32)
#define MODE_PERHEAD_ROW 1   // Cb[b][h][s][dh] = bf16(acc + bias)  (q, k)
#define MODE_PERHEAD_COL 2   // Cb[b][h][dh][s] = bf16(acc + bias)  (v)

// ---------------------------------------------------------------------------
// GEMM: C[M,N] = A[M,K] @ Bt[N,K]^T + bias[N].  A,Bt bf16 row-major.
// Block 256 thr = 8 waves (2x4). Tile 128x128x64, double-buffered async LDS.
// ---------------------------------------------------------------------------
__global__ __launch_bounds__(256) void gemm_bf16_kernel(
    const ushort_t* __restrict__ A, const ushort_t* __restrict__ Bt,
    const float* __restrict__ bias, float* __restrict__ Cf,
    ushort_t* __restrict__ Cb, int M, int K, int N, int mode, int skv) {
  __shared__ __align__(16) ushort_t sA[2][128 * 72];   // [m][k], pad->72
  __shared__ __align__(16) ushort_t sB[2][128 * 72];   // [n][k]

  const int tid  = threadIdx.x;
  const int lane = tid & 31;
  const int wave = tid >> 5;
  const int wm   = wave >> 2;            // 0..1
  const int wn   = wave & 3;             // 0..3
  const int l16  = lane & 15;
  const int half = lane >> 4;
  const int kb0  = half * 8;             // per-ISA 16-bit A/B frag K offset

  const int m0 = blockIdx.y * 128;
  const int n0 = blockIdx.x * 128;
  const ushort_t* Abase = A  + (size_t)m0 * K;
  const ushort_t* Bbase = Bt + (size_t)n0 * K;

  // copy-chunk assignment: 1024 chunks of 16B per tile; 4 per thread per tile
  const int c_row = (tid * 4) >> 3;        // tid*4 .. tid*4+3 are consecutive
  // chunk c = tid + j*256 -> row = c>>3, c8 = c&7  (computed in loop)

  v8f zerov = {};
  v8f acc[4][2];
  #pragma unroll
  for (int i = 0; i < 4; ++i)
    #pragma unroll
    for (int j = 0; j < 2; ++j) acc[i][j] = zerov;
  (void)c_row;

  const int KT = K / 64;                   // 16 tiles

  // issue async copies for tile t into buffer buf
  auto issue = [&](int t, int buf) {
    const int k0 = t * 64;
    #pragma unroll
    for (int j = 0; j < 4; ++j) {
      int c = tid + j * 256;               // 0..1023
      int row = c >> 3, c8 = c & 7;
      async_cp16((unsigned)(uintptr_t)&sA[buf][row * 72 + c8 * 8],
                 Abase + (size_t)row * K + k0 + c8 * 8);
    }
    #pragma unroll
    for (int j = 0; j < 4; ++j) {
      int c = tid + j * 256;
      int row = c >> 3, c8 = c & 7;
      async_cp16((unsigned)(uintptr_t)&sB[buf][row * 72 + c8 * 8],
                 Bbase + (size_t)row * K + k0 + c8 * 8);
    }
  };

  issue(0, 0);
  for (int t = 0; t < KT; ++t) {
    const int buf = t & 1;
    wait_async0();                 // tile t resident in LDS (issued last iter)
    __syncthreads();               // visible to all waves; prev buf free
    if (t + 1 < KT) issue(t + 1, buf ^ 1);   // overlap with compute below

    #pragma unroll
    for (int ks = 0; ks < 2; ++ks) {         // two 32-wide WMMA k-steps
      FragAB af[4], bf[2];
      #pragma unroll
      for (int mf = 0; mf < 4; ++mf) {
        const ushort_t* p = &sA[buf][(wm * 64 + mf * 16 + l16) * 72 + ks * 32];
        af[mf].q[0] = *(const uint4*)(p + kb0);
        af[mf].q[1] = *(const uint4*)(p + 16 + kb0);
      }
      #pragma unroll
      for (int nf = 0; nf < 2; ++nf) {
        const ushort_t* p = &sB[buf][(wn * 32 + nf * 16 + l16) * 72 + ks * 32];
        bf[nf].q[0] = *(const uint4*)(p + kb0);
        bf[nf].q[1] = *(const uint4*)(p + 16 + kb0);
      }
      #pragma unroll
      for (int mf = 0; mf < 4; ++mf)
        #pragma unroll
        for (int nf = 0; nf < 2; ++nf)
          acc[mf][nf] = __builtin_amdgcn_wmma_f32_16x16x32_bf16(
              false, af[mf].v, false, bf[nf].v, (short)0, acc[mf][nf], false, false);
    }
    __syncthreads();               // done reading buf before next overwrite
  }

  // ---- epilogue ----
  const int bb     = m0 / skv;             // batch index (128 | skv)
  const int s_base = m0 - bb * skv;        // token within batch
  #pragma unroll
  for (int mf = 0; mf < 4; ++mf) {
    const int mrow = wm * 64 + mf * 16 + 8 * half;     // + r
    #pragma unroll
    for (int nf = 0; nf < 2; ++nf) {
      const int n = n0 + wn * 32 + nf * 16 + l16;
      const float bv = bias[n];
      if (mode == MODE_F32) {
        float* cp = Cf + (size_t)(m0 + mrow) * N + n;
        #pragma unroll
        for (int r = 0; r < 8; ++r) cp[(size_t)r * N] = acc[mf][nf][r] + bv;
      } else {
        const int hh = n >> 6, dd = n & 63;
        const int srow = s_base + mrow;
        if (mode == MODE_PERHEAD_ROW) {
          ushort_t* cp = Cb + ((size_t)(bb * NHEAD + hh) * skv + srow) * DH + dd;
          #pragma unroll
          for (int r = 0; r < 8; ++r) cp[(size_t)r * DH] = f2bf(acc[mf][nf][r] + bv);
        } else {
          ushort_t* cp = Cb + ((size_t)(bb * NHEAD + hh) * DH + dd) * skv + srow;
          #pragma unroll
          for (int r = 0; r < 8; ++r) cp[r] = f2bf(acc[mf][nf][r] + bv);
        }
      }
    }
  }
}

// ---------------------------------------------------------------------------
// Fused flash attention. Block = 4 waves = 64 q-rows of one (b,h).
// qb,kb: bf16 [B][H][S][64] ; vb: bf16 [B][H][64][S] ; out: bf16 [B*S][1024].
// ---------------------------------------------------------------------------
__global__ __launch_bounds__(128) void attention_kernel(
    const ushort_t* __restrict__ Qb, const ushort_t* __restrict__ Kb,
    const ushort_t* __restrict__ Vb, ushort_t* __restrict__ Ob,
    int Skv, int causal) {
  __shared__ __align__(16) ushort_t sQ[64 * 72];   // [qrow][dh]
  __shared__ __align__(16) ushort_t sK[64 * 72];   // [krow][dh]
  __shared__ __align__(16) ushort_t sVt[64 * 72];  // [dh][krow]
  __shared__ __align__(16) ushort_t sP[64 * 72];   // [qrow][krow]

  const int tid  = threadIdx.x;
  const int lane = tid & 31;
  const int wave = tid >> 5;
  const int l16  = lane & 15;
  const int half = lane >> 4;
  const int kb0  = half * 8;

  const int qt = blockIdx.x;
  const int bh = blockIdx.y;
  const int b  = bh / NHEAD, hh = bh % NHEAD;

  const ushort_t* qp = Qb + (size_t)bh * SEQ * DH;
  const ushort_t* kp = Kb + (size_t)bh * Skv * DH;
  const ushort_t* vp = Vb + (size_t)bh * DH * Skv;

  #pragma unroll
  for (int j = 0; j < 4; ++j) {            // Q tile: 512 x 16B chunks
    int c = tid + j * 128;
    int row = c >> 3, c8 = c & 7;
    *(uint4*)&sQ[row * 72 + c8 * 8] =
        *(const uint4*)(qp + (size_t)(qt * 64 + row) * DH + c8 * 8);
  }

  v8f zerov = {};
  v8f o_acc[4];
  #pragma unroll
  for (int i = 0; i < 4; ++i) o_acc[i] = zerov;
  float mrow[8], lrow[8];
  #pragma unroll
  for (int r = 0; r < 8; ++r) { mrow[r] = -3.0e38f; lrow[r] = 0.0f; }

  const int ktiles = causal ? (qt + 1) : (Skv / 64);
  for (int kt = 0; kt < ktiles; ++kt) {
    __syncthreads();
    #pragma unroll
    for (int j = 0; j < 4; ++j) {          // K tile (row-major per head)
      int c = tid + j * 128;
      int row = c >> 3, c8 = c & 7;
      *(uint4*)&sK[row * 72 + c8 * 8] =
          *(const uint4*)(kp + (size_t)(kt * 64 + row) * DH + c8 * 8);
    }
    #pragma unroll
    for (int j = 0; j < 4; ++j) {          // V tile (already [dh][s])
      int c = tid + j * 128;
      int row = c >> 3, c8 = c & 7;        // row = dh
      *(uint4*)&sVt[row * 72 + c8 * 8] =
          *(const uint4*)(vp + (size_t)row * Skv + kt * 64 + c8 * 8);
    }
    __syncthreads();

    // ---- scores: S = Q @ K^T (dh=64 -> two k-steps) ----
    FragAB aq0, aq1;
    {
      const ushort_t* p = &sQ[(wave * 16 + l16) * 72];
      aq0.q[0] = *(const uint4*)(p + kb0);       aq0.q[1] = *(const uint4*)(p + 16 + kb0);
      aq1.q[0] = *(const uint4*)(p + 32 + kb0);  aq1.q[1] = *(const uint4*)(p + 48 + kb0);
    }
    v8f sc[4];
    #pragma unroll
    for (int nf = 0; nf < 4; ++nf) {
      const ushort_t* p = &sK[(nf * 16 + l16) * 72];
      FragAB b0, b1;
      b0.q[0] = *(const uint4*)(p + kb0);       b0.q[1] = *(const uint4*)(p + 16 + kb0);
      b1.q[0] = *(const uint4*)(p + 32 + kb0);  b1.q[1] = *(const uint4*)(p + 48 + kb0);
      v8f z = zerov;
      z = __builtin_amdgcn_wmma_f32_16x16x32_bf16(false, aq0.v, false, b0.v, (short)0, z, false, false);
      z = __builtin_amdgcn_wmma_f32_16x16x32_bf16(false, aq1.v, false, b1.v, (short)0, z, false, false);
      sc[nf] = z;
    }

    // ---- scale + causal mask ----
    #pragma unroll
    for (int nf = 0; nf < 4; ++nf) {
      int kg = kt * 64 + nf * 16 + l16;
      #pragma unroll
      for (int r = 0; r < 8; ++r) {
        float sv = sc[nf][r] * 0.125f;           // 1/sqrt(64)
        if (causal) {
          int qg = qt * 64 + wave * 16 + r + 8 * half;
          if (kg > qg) sv -= 1.0e9f;
        }
        sc[nf][r] = sv;
      }
    }

    // ---- online softmax (row = r + 8*half, replicated across 16 lanes) ----
    #pragma unroll
    for (int r = 0; r < 8; ++r) {
      float rmax = fmaxf(fmaxf(sc[0][r], sc[1][r]), fmaxf(sc[2][r], sc[3][r]));
      #pragma unroll
      for (int off = 8; off >= 1; off >>= 1)
        rmax = fmaxf(rmax, __shfl_xor(rmax, off, 32));
      float mn = fmaxf(mrow[r], rmax);
      float corr = __expf(mrow[r] - mn);
      mrow[r] = mn;
      lrow[r] *= corr;
      #pragma unroll
      for (int onf = 0; onf < 4; ++onf) o_acc[onf][r] *= corr;
      float rs = 0.0f;
      int prow = wave * 16 + r + 8 * half;
      #pragma unroll
      for (int nf = 0; nf < 4; ++nf) {
        float pv = __expf(sc[nf][r] - mn);
        rs += pv;
        sP[prow * 72 + nf * 16 + l16] = f2bf(pv);
      }
      #pragma unroll
      for (int off = 8; off >= 1; off >>= 1) rs += __shfl_xor(rs, off, 32);
      lrow[r] += rs;
    }
    __syncthreads();

    // ---- O += P @ V (keys=64 -> two k-steps) ----
    FragAB ap0, ap1;
    {
      const ushort_t* p = &sP[(wave * 16 + l16) * 72];
      ap0.q[0] = *(const uint4*)(p + kb0);       ap0.q[1] = *(const uint4*)(p + 16 + kb0);
      ap1.q[0] = *(const uint4*)(p + 32 + kb0);  ap1.q[1] = *(const uint4*)(p + 48 + kb0);
    }
    #pragma unroll
    for (int onf = 0; onf < 4; ++onf) {
      const ushort_t* p = &sVt[(onf * 16 + l16) * 72];
      FragAB b0, b1;
      b0.q[0] = *(const uint4*)(p + kb0);       b0.q[1] = *(const uint4*)(p + 16 + kb0);
      b1.q[0] = *(const uint4*)(p + 32 + kb0);  b1.q[1] = *(const uint4*)(p + 48 + kb0);
      o_acc[onf] = __builtin_amdgcn_wmma_f32_16x16x32_bf16(false, ap0.v, false, b0.v, (short)0, o_acc[onf], false, false);
      o_acc[onf] = __builtin_amdgcn_wmma_f32_16x16x32_bf16(false, ap1.v, false, b1.v, (short)0, o_acc[onf], false, false);
    }
  }

  // ctx out: bf16 row-major [B*S][D]
  ushort_t* op = Ob + ((size_t)b * SEQ) * D_MODEL + hh * DH;
  #pragma unroll
  for (int onf = 0; onf < 4; ++onf)
    #pragma unroll
    for (int r = 0; r < 8; ++r) {
      int qg = qt * 64 + wave * 16 + r + 8 * half;
      op[(size_t)qg * D_MODEL + onf * 16 + l16] = f2bf(o_acc[onf][r] / lrow[r]);
    }
}

// ---------------------------------------------------------------------------
// out = LayerNorm(x + y) * g + b ; also writes bf16 shadow.
// ---------------------------------------------------------------------------
static __device__ __forceinline__ float block_sum(float v) {
  __shared__ float red[8];
  #pragma unroll
  for (int off = 16; off >= 1; off >>= 1) v += __shfl_xor(v, off, 32);
  if ((threadIdx.x & 31) == 0) red[threadIdx.x >> 5] = v;
  __syncthreads();
  float tot = 0.0f;
  #pragma unroll
  for (int i = 0; i < 8; ++i) tot += red[i];
  __syncthreads();
  return tot;
}

__global__ __launch_bounds__(256) void add_ln_kernel(
    const float* __restrict__ X, const float* __restrict__ Y,
    const float* __restrict__ g, const float* __restrict__ bta,
    float* __restrict__ Out, ushort_t* __restrict__ Outb) {
  const size_t row = blockIdx.x;
  float v[4];
  float s = 0.0f;
  #pragma unroll
  for (int i = 0; i < 4; ++i) {
    int c = threadIdx.x + i * 256;
    v[i] = X[row * D_MODEL + c] + Y[row * D_MODEL + c];
    s += v[i];
  }
  float mu = block_sum(s) * (1.0f / D_MODEL);
  float vs = 0.0f;
  #pragma unroll
  for (int i = 0; i < 4; ++i) { float d = v[i] - mu; vs += d * d; }
  float var = block_sum(vs) * (1.0f / D_MODEL);
  float rstd = rsqrtf(var + 1e-6f);
  #pragma unroll
  for (int i = 0; i < 4; ++i) {
    int c = threadIdx.x + i * 256;
    float r = (v[i] - mu) * rstd * g[c] + bta[c];
    Out[row * D_MODEL + c] = r;
    Outb[row * D_MODEL + c] = f2bf(r);
  }
}

// ---------------------------------------------------------------------------
// h = emb[x]*sqrt(D) + pos_embedding (concat sin/cos, faithful to reference).
// ---------------------------------------------------------------------------
__global__ __launch_bounds__(256) void embed_kernel(
    const int* __restrict__ x, const float* __restrict__ emb,
    float* __restrict__ h, ushort_t* __restrict__ hb) {
  const int row = blockIdx.x;
  const int s = row % SEQ;
  const int tok = x[row];
  const float LN1E4_O512 = 9.2103403719761836f / 512.0f;
  #pragma unroll
  for (int i = 0; i < 4; ++i) {
    int c = threadIdx.x + i * 256;
    float e = emb[(size_t)tok * D_MODEL + c] * 32.0f;     // sqrt(1024)
    float pe;
    if (c < 512) pe = __sinf((float)s * __expf((float)c * LN1E4_O512));
    else         pe = __cosf((float)s * __expf((float)(c - 512) * LN1E4_O512));
    float r = e + pe;
    h[(size_t)row * D_MODEL + c] = r;
    hb[(size_t)row * D_MODEL + c] = f2bf(r);
  }
}

// f32 -> bf16 bulk convert (for encoder_outputs)
__global__ __launch_bounds__(256) void cvt_bf16_kernel(
    const float* __restrict__ src, ushort_t* __restrict__ dst) {
  const size_t base = (size_t)blockIdx.x * 1024;
  #pragma unroll
  for (int i = 0; i < 4; ++i) {
    size_t idx = base + threadIdx.x + i * 256;
    dst[idx] = f2bf(src[idx]);
  }
}

// Transpose+convert one DxD f32 matrix per blockIdx.z into bf16 [N][K].
__global__ __launch_bounds__(256) void wt_transpose_kernel(
    const float* __restrict__ src, ushort_t* __restrict__ dst) {
  __shared__ float tile[32][33];
  const size_t DD = (size_t)D_MODEL * D_MODEL;
  const float* S = src + (size_t)blockIdx.z * DD;
  ushort_t*    T = dst + (size_t)blockIdx.z * DD;
  const int n0 = blockIdx.x * 32, k0 = blockIdx.y * 32;
  const int tx = threadIdx.x & 31, ty = threadIdx.x >> 5;   // 32 x 8
  #pragma unroll
  for (int j = 0; j < 4; ++j)
    tile[ty + j * 8][tx] = S[(size_t)(k0 + ty + j * 8) * D_MODEL + n0 + tx];
  __syncthreads();
  #pragma unroll
  for (int j = 0; j < 4; ++j)
    T[(size_t)(n0 + ty + j * 8) * D_MODEL + k0 + tx] = f2bf(tile[tx][ty + j * 8]);
}

// ---------------------------------------------------------------------------
extern "C" void kernel_launch(void* const* d_in, const int* in_sizes, int n_in,
                              void* d_out, int out_size, void* d_ws, size_t ws_size,
                              hipStream_t stream) {
  const int*   x    = (const int*)d_in[0];
  const float* enc  = (const float*)d_in[1];
  const float* emb  = (const float*)d_in[4];
  const float* sa_w = (const float*)d_in[5];
  const float* sa_b = (const float*)d_in[6];
  const float* ca_w = (const float*)d_in[7];
  const float* ca_b = (const float*)d_in[8];
  const float* ff_w = (const float*)d_in[9];
  const float* ff_b = (const float*)d_in[10];
  const float* ln_g = (const float*)d_in[11];
  const float* ln_b = (const float*)d_in[12];
  float* out = (float*)d_out;

  const size_t BUF = (size_t)NTOK * D_MODEL;            // 4M elements
  const size_t DD  = (size_t)D_MODEL * D_MODEL;
  char* w = (char*)d_ws;
  float*    h    = (float*)w;                 w += BUF * 4;
  float*    t    = (float*)w;                 w += BUF * 4;
  ushort_t* hb   = (ushort_t*)w;              w += BUF * 2;
  ushort_t* encb = (ushort_t*)w;              w += BUF * 2;
  ushort_t* qb   = (ushort_t*)w;              w += BUF * 2;
  ushort_t* kb   = (ushort_t*)w;              w += BUF * 2;
  ushort_t* vb   = (ushort_t*)w;              w += BUF * 2;
  ushort_t* ctxb = (ushort_t*)w;              w += BUF * 2;
  ushort_t* wt   = (ushort_t*)w;              // 36 * DD bf16 = 72 MB

  const dim3 gGemm(D_MODEL / 128, NTOK / 128);   // (8, 32)
  const dim3 gAttn(SEQ / 64, BATCH * NHEAD);     // (32, 32)
  const dim3 gT(D_MODEL / 32, D_MODEL / 32);

  // one-time conversions (per launch)
  wt_transpose_kernel<<<dim3(gT.x, gT.y, NLAYER * 4), 256, 0, stream>>>(sa_w, wt);
  wt_transpose_kernel<<<dim3(gT.x, gT.y, NLAYER * 4), 256, 0, stream>>>(ca_w, wt + 16 * DD);
  wt_transpose_kernel<<<dim3(gT.x, gT.y, NLAYER),     256, 0, stream>>>(ff_w, wt + 32 * DD);
  cvt_bf16_kernel<<<NTOK, 256, 0, stream>>>(enc, encb);
  embed_kernel<<<NTOK, 256, 0, stream>>>(x, emb, h, hb);

  for (int l = 0; l < NLAYER; ++l) {
    const ushort_t* Wsa = wt + (size_t)l * 4 * DD;
    const ushort_t* Wca = wt + (16 + (size_t)l * 4) * DD;
    const float* Bsa = sa_b + (size_t)l * 4 * D_MODEL;
    const float* Bca = ca_b + (size_t)l * 4 * D_MODEL;

    // ---- self-attention ----
    gemm_bf16_kernel<<<gGemm, 256, 0, stream>>>(hb, Wsa + 0 * DD, Bsa + 0 * D_MODEL,
        nullptr, qb, NTOK, D_MODEL, D_MODEL, MODE_PERHEAD_ROW, SEQ);
    gemm_bf16_kernel<<<gGemm, 256, 0, stream>>>(hb, Wsa + 1 * DD, Bsa + 1 * D_MODEL,
        nullptr, kb, NTOK, D_MODEL, D_MODEL, MODE_PERHEAD_ROW, SEQ);
    gemm_bf16_kernel<<<gGemm, 256, 0, stream>>>(hb, Wsa + 2 * DD, Bsa + 2 * D_MODEL,
        nullptr, vb, NTOK, D_MODEL, D_MODEL, MODE_PERHEAD_COL, SEQ);
    attention_kernel<<<gAttn, 128, 0, stream>>>(qb, kb, vb, ctxb, SEQ, 1);
    gemm_bf16_kernel<<<gGemm, 256, 0, stream>>>(ctxb, Wsa + 3 * DD, Bsa + 3 * D_MODEL,
        t, nullptr, NTOK, D_MODEL, D_MODEL, MODE_F32, SEQ);
    add_ln_kernel<<<NTOK, 256, 0, stream>>>(h, t, ln_g + (size_t)(l * 3 + 0) * D_MODEL,
        ln_b + (size_t)(l * 3 + 0) * D_MODEL, h, hb);

    // ---- cross-attention ----
    gemm_bf16_kernel<<<gGemm, 256, 0, stream>>>(hb, Wca + 0 * DD, Bca + 0 * D_MODEL,
        nullptr, qb, NTOK, D_MODEL, D_MODEL, MODE_PERHEAD_ROW, SEQ);
    gemm_bf16_kernel<<<gGemm, 256, 0, stream>>>(encb, Wca + 1 * DD, Bca + 1 * D_MODEL,
        nullptr, kb, BATCH * SENC, D_MODEL, D_MODEL, MODE_PERHEAD_ROW, SENC);
    gemm_bf16_kernel<<<gGemm, 256, 0, stream>>>(encb, Wca + 2 * DD, Bca + 2 * D_MODEL,
        nullptr, vb, BATCH * SENC, D_MODEL, D_MODEL, MODE_PERHEAD_COL, SENC);
    attention_kernel<<<gAttn, 128, 0, stream>>>(qb, kb, vb, ctxb, SENC, 0);
    gemm_bf16_kernel<<<gGemm, 256, 0, stream>>>(ctxb, Wca + 3 * DD, Bca + 3 * D_MODEL,
        t, nullptr, NTOK, D_MODEL, D_MODEL, MODE_F32, SEQ);
    add_ln_kernel<<<NTOK, 256, 0, stream>>>(h, t, ln_g + (size_t)(l * 3 + 1) * D_MODEL,
        ln_b + (size_t)(l * 3 + 1) * D_MODEL, h, hb);

    // ---- FFN (plain Dense, no activation) ----
    gemm_bf16_kernel<<<gGemm, 256, 0, stream>>>(hb, wt + (32 + (size_t)l) * DD,
        ff_b + (size_t)l * D_MODEL, t, nullptr, NTOK, D_MODEL, D_MODEL, MODE_F32, SEQ);
    float* dst = (l == NLAYER - 1) ? out : h;
    add_ln_kernel<<<NTOK, 256, 0, stream>>>(h, t, ln_g + (size_t)(l * 3 + 2) * D_MODEL,
        ln_b + (size_t)(l * 3 + 2) * D_MODEL, dst, hb);
  }
}
